// CPPA_34703335752239
// MI455X (gfx1250) — compile-verified, matrix operand order checked by reading, not run on hardware
//
#include <hip/hip_runtime.h>
#include <hip/hip_bf16.h>

typedef __attribute__((ext_vector_type(16))) __bf16 v16bf;
typedef __attribute__((ext_vector_type(8)))  float  v8f;

#define DEV __device__ __forceinline__

static constexpr int Bb = 4, Ntok = 8192, C = 256, H = 4;
static constexpr int Mrows = Bb * Ntok;   // 32768

// ---------- bf16 helpers (bf16 stored as raw ushort in workspace) ----------
DEV unsigned short f2bf_bits(float f) {
  union { __bf16 b; unsigned short s; } c; c.b = (__bf16)f; return c.s;
}
DEV float bf2f(unsigned short h) {
  union { unsigned u; float f; } a; a.u = ((unsigned)h) << 16; return a.f;
}

union U8x16 { uint4 u; unsigned short s[8]; };

DEV uint4 pack8(const float* f) {
  U8x16 c;
#pragma unroll
  for (int j = 0; j < 8; ++j) c.s[j] = f2bf_bits(f[j]);
  return c.u;
}

// ---------- async global -> LDS copy (GLOBAL_LOAD_ASYNC_TO_LDS_B128) -------
// Per-lane: LDS[ldsAddr] = MEM[gaddr], 16 bytes, tracked by ASYNCcnt.
DEV void async_b128(void* lds, const void* g) {
  const unsigned ldsAddr = (unsigned)(unsigned long long)lds;  // LDS addr = low 32 bits
  asm volatile("global_load_async_to_lds_b128 %0, %1, off"
               :: "v"(ldsAddr), "v"(g) : "memory");
}
DEV void wait_async() {
  asm volatile("s_wait_asynccnt 0x0" ::: "memory");
}

// Packed-B tile layout: 32x16 (KxN) tile stored lane-major [32][16]:
// element (kk,n16) lives at L*16 + j, L = n16 + ((kk>>3)&1)*16,
// j = ((kk>>4)<<3)|(kk&7). Fragment load = 2 aligned ds_load_b128.
DEV int bpk(int kk, int n16) {
  return (n16 + ((kk >> 3) & 1) * 16) * 16 + (((kk >> 4) << 3) | (kk & 7));
}

// ---------- WMMA fragment loads ----------
DEV v16bf load_a(const unsigned short* src, int ld) {
  const int lane = threadIdx.x & 31;
  const int m = lane & 15, half = lane >> 4;
  const unsigned short* r = src + m * ld + half * 8;
  union { uint4 u[2]; v16bf v; } c;
  c.u[0] = *(const uint4*)(r);
  c.u[1] = *(const uint4*)(r + 16);
  return c.v;
}
DEV v16bf load_bp(const unsigned short* tile) {
  const int lane = threadIdx.x & 31;
  const uint4* p = (const uint4*)(tile + lane * 16);
  union { uint4 u[2]; v16bf v; } c;
  c.u[0] = p[0];
  c.u[1] = p[1];
  return c.v;
}

DEV v8f wmma_bf16(v16bf a, v16bf b, v8f c) {
  return __builtin_amdgcn_wmma_f32_16x16x32_bf16(
      false, a, false, b, (short)0, c, false, false);
}

// =====================================================================
// K1: Y[M,Cout](bf16) = X[M,256](f32) @ W[Cout,256](f32)^T
// =====================================================================
__global__ void __launch_bounds__(256) k_gemm_f32_bf16(
    const float* __restrict__ X, const float* __restrict__ W,
    unsigned short* __restrict__ Y, int Cout) {
  __shared__ __align__(16) unsigned short Xs[128 * 32];
  __shared__ __align__(16) unsigned short Wp[4 * 512];
  const int t = threadIdx.x, wv = t >> 5, lane = t & 31;
  const int gm0 = blockIdx.x * 128, gn0 = blockIdx.y * 64;
  const int mrow = (wv & 3) * 32, ncol = (wv >> 2) * 32;
  const int xr = t >> 1, xc = (t & 1) * 16;
  const int wn = t >> 2, wk = (t & 3) * 8;
  v8f acc[2][2] = {};
  for (int k0 = 0; k0 < 256; k0 += 32) {
    __syncthreads();
    {
      float f[16];
      const float* gx = X + (size_t)(gm0 + xr) * C + k0 + xc;
#pragma unroll
      for (int j = 0; j < 16; ++j) f[j] = gx[j];
      uint4* dst = (uint4*)&Xs[xr * 32 + xc];
      dst[0] = pack8(f);
      dst[1] = pack8(f + 8);
    }
    {
      float g[8];
      const float* gw = W + (size_t)(gn0 + wn) * C + k0 + wk;
#pragma unroll
      for (int j = 0; j < 8; ++j) g[j] = gw[j];
      const int L = (wn & 15) + ((wk >> 3) & 1) * 16;
      const int j0 = (wk >> 4) << 3;
      *(uint4*)&Wp[(wn >> 4) * 512 + L * 16 + j0] = pack8(g);
    }
    if (k0 + 32 < 256)
      __builtin_prefetch(X + (size_t)(gm0 + xr) * C + k0 + 32 + xc, 0, 1);
    __syncthreads();
    v16bf a0 = load_a(&Xs[mrow * 32], 32);
    v16bf a1 = load_a(&Xs[(mrow + 16) * 32], 32);
    v16bf b0 = load_bp(&Wp[(ncol >> 4) * 512]);
    v16bf b1 = load_bp(&Wp[(ncol >> 4) * 512 + 512]);
    acc[0][0] = wmma_bf16(a0, b0, acc[0][0]);
    acc[0][1] = wmma_bf16(a0, b1, acc[0][1]);
    acc[1][0] = wmma_bf16(a1, b0, acc[1][0]);
    acc[1][1] = wmma_bf16(a1, b1, acc[1][1]);
  }
  const int lr = (lane >> 4) * 8, lc = lane & 15;
#pragma unroll
  for (int tm = 0; tm < 2; ++tm)
#pragma unroll
    for (int tn = 0; tn < 2; ++tn)
#pragma unroll
      for (int v = 0; v < 8; ++v)
        Y[(size_t)(gm0 + mrow + tm * 16 + lr + v) * Cout +
          (gn0 + ncol + tn * 16 + lc)] = f2bf_bits(acc[tm][tn][v]);
}

// =====================================================================
// K2: inv-norms over token dim for Q and K
// =====================================================================
__global__ void __launch_bounds__(256) k_norms(
    const unsigned short* __restrict__ Q, const unsigned short* __restrict__ K,
    float* __restrict__ invq, float* __restrict__ invk) {
  const int b = blockIdx.x, c = threadIdx.x;
  const size_t base = (size_t)b * Ntok * C + c;
  float sq = 0.f, sk = 0.f;
  for (int n = 0; n < Ntok; ++n) {
    float a = bf2f(Q[base + (size_t)n * C]);
    float d = bf2f(K[base + (size_t)n * C]);
    sq += a * a;
    sk += d * d;
  }
  invq[b * C + c] = 1.f / fmaxf(sqrtf(sq), 1e-12f);
  invk[b * C + c] = 1.f / fmaxf(sqrtf(sk), 1e-12f);
}

// =====================================================================
// K3: per (b,h): G = Q^T K -> scaled softmax -> attn (packed global),
//     KP = K^T Ew^T (+E_b)*invq*temp2 -> kp2 (packed), VP -> vp (packed)
// =====================================================================
__global__ void __launch_bounds__(256) k_proj_attn(
    const unsigned short* __restrict__ Qb, const unsigned short* __restrict__ Kb,
    const unsigned short* __restrict__ Vsab,
    const float* __restrict__ E_w, const float* __restrict__ E_b,
    const float* __restrict__ invq, const float* __restrict__ invk,
    const float* __restrict__ temp, const float* __restrict__ temp2,
    unsigned short* __restrict__ attn_pk, unsigned short* __restrict__ kp2,
    unsigned short* __restrict__ vp) {
  __shared__ __align__(16) unsigned short QsT[64 * 32];  // [d][n]
  __shared__ __align__(16) unsigned short KsT[64 * 32];
  __shared__ __align__(16) unsigned short VsT[64 * 32];
  __shared__ __align__(16) unsigned short Kpk[4 * 512];  // B tiles (k=n, col=e)
  __shared__ __align__(16) unsigned short Epk[4 * 512];  // B tiles (k=n, col=p)
  __shared__ float Gs[64 * 64];
  const int t = threadIdx.x, wv = t >> 5, lane = t & 31;
  const int bh = blockIdx.x, b = bh >> 2, h = bh & 3;
  const size_t base = (size_t)b * Ntok * C + h * 64;
  const int sn = t >> 3, sd8 = (t & 7) * 8;     // Q/K/V staging
  const int sp = t >> 2, sk8 = (t & 3) * 8;     // E staging
  v8f acc[6] = {};
  for (int n0 = 0; n0 < Ntok; n0 += 32) {
    __syncthreads();
    {
      const size_t g = base + (size_t)(n0 + sn) * C + sd8;
      U8x16 uq, uk, uvv;
      uq.u = *(const uint4*)(Qb + g);
      uk.u = *(const uint4*)(Kb + g);
      uvv.u = *(const uint4*)(Vsab + g);
#pragma unroll
      for (int j = 0; j < 8; ++j) {
        const int d = sd8 + j;
        QsT[d * 32 + sn] = uq.s[j];
        KsT[d * 32 + sn] = uk.s[j];
        VsT[d * 32 + sn] = uvv.s[j];
        Kpk[(d >> 4) * 512 + bpk(sn, d & 15)] = uk.s[j];
      }
      float ef[8];
      const float* ge = E_w + (size_t)sp * Ntok + n0 + sk8;
#pragma unroll
      for (int j = 0; j < 8; ++j) ef[j] = ge[j];
      const int L = (sp & 15) + ((sk8 >> 3) & 1) * 16;
      const int j0 = (sk8 >> 4) << 3;
      *(uint4*)&Epk[(sp >> 4) * 512 + L * 16 + j0] = pack8(ef);
    }
    __syncthreads();
#pragma unroll
    for (int j = 0; j < 6; ++j) {
      const int id = wv * 6 + j;
      const int outk = id >> 4, tt = id & 15, tm = tt >> 2, tn = tt & 3;
      const unsigned short* Asrc = (outk == 0) ? QsT : (outk == 1) ? KsT : VsT;
      const unsigned short* Bsrc = (outk == 0) ? Kpk : Epk;
      v16bf a = load_a(&Asrc[tm * 16 * 32], 32);
      v16bf bmat = load_bp(&Bsrc[tn * 512]);
      acc[j] = wmma_bf16(a, bmat, acc[j]);
    }
  }
  __syncthreads();
  const float tq = temp[h], t2 = temp2[h];
  const int lr = (lane >> 4) * 8, lc = lane & 15;
#pragma unroll
  for (int j = 0; j < 6; ++j) {
    const int id = wv * 6 + j;
    const int outk = id >> 4, tt = id & 15, tm = tt >> 2, tn = tt & 3;
#pragma unroll
    for (int v = 0; v < 8; ++v) {
      const int d = tm * 16 + lr + v, e = tn * 16 + lc;  // e doubles as p
      const float val = acc[j][v];
      if (outk == 0) {
        Gs[d * 64 + e] = val * invq[b * C + h * 64 + d] *
                         invk[b * C + h * 64 + e] * tq;
      } else if (outk == 1) {
        float kv = (val + E_b[e]) * invq[b * C + h * 64 + d] * t2;
        kp2[(size_t)bh * 4096 + (d >> 5) * 2048 + tn * 512 +
            bpk(d & 31, e & 15)] = f2bf_bits(kv);
      } else {
        vp[(size_t)bh * 4096 + (e >> 5) * 2048 + tm * 512 +
           bpk(e & 31, d & 15)] = f2bf_bits(val + E_b[e]);
      }
    }
  }
  __syncthreads();
  if (t < 64) {   // channel-attention softmax over e; row d = t
    float mx = -3.4e38f;
    for (int e = 0; e < 64; ++e) mx = fmaxf(mx, Gs[t * 64 + e]);
    float s = 0.f;
    for (int e = 0; e < 64; ++e) s += __expf(Gs[t * 64 + e] - mx);
    const float inv = 1.f / s;
    for (int e = 0; e < 64; ++e)
      attn_pk[(size_t)bh * 4096 + (e >> 5) * 2048 + (t >> 4) * 512 +
              bpk(e & 31, t & 15)] = f2bf_bits(__expf(Gs[t * 64 + e] - mx) * inv);
  }
}

// =====================================================================
// K4: x_ca[n, h*64+d] = sum_e Vca[n, h*64+e] * attn[d][e]
// =====================================================================
__global__ void __launch_bounds__(256) k_ca_apply(
    const unsigned short* __restrict__ Vca,
    const unsigned short* __restrict__ attn_pk,
    unsigned short* __restrict__ xca) {
  __shared__ __align__(16) unsigned short Bp[2 * 4 * 512];
  __shared__ __align__(16) unsigned short Vst[128 * 64];
  const int t = threadIdx.x, wv = t >> 5, lane = t & 31;
  const int bh = blockIdx.y, b = bh >> 2, h = bh & 3;
  const int n0 = blockIdx.x * 128;
  const size_t base = (size_t)b * Ntok * C + h * 64;
  {  // async packed copy of attn (4096 ushorts) + V tile (8192 ushorts)
    const unsigned short* src = attn_pk + (size_t)bh * 4096;
    async_b128(&Bp[t * 8], src + t * 8);
    async_b128(&Bp[2048 + t * 8], src + 2048 + t * 8);
    const int r = t >> 1, c32 = (t & 1) * 32;
    const unsigned short* vsrc = Vca + base + (size_t)(n0 + r) * C + c32;
#pragma unroll
    for (int j = 0; j < 4; ++j)
      async_b128(&Vst[r * 64 + c32 + j * 8], vsrc + j * 8);
  }
  wait_async();
  __syncthreads();
  const int mrow = (wv & 3) * 32, dcol = (wv >> 2) * 32;
  v8f acc[2][2] = {};
#pragma unroll
  for (int k0 = 0; k0 < 64; k0 += 32) {
    v16bf a0 = load_a(&Vst[mrow * 64 + k0], 64);
    v16bf a1 = load_a(&Vst[(mrow + 16) * 64 + k0], 64);
    const unsigned short* bb = &Bp[(k0 >> 5) * 2048 + (dcol >> 4) * 512];
    v16bf b0 = load_bp(bb);
    v16bf b1 = load_bp(bb + 512);
    acc[0][0] = wmma_bf16(a0, b0, acc[0][0]);
    acc[0][1] = wmma_bf16(a0, b1, acc[0][1]);
    acc[1][0] = wmma_bf16(a1, b0, acc[1][0]);
    acc[1][1] = wmma_bf16(a1, b1, acc[1][1]);
  }
  const int lr = (lane >> 4) * 8, lc = lane & 15;
#pragma unroll
  for (int tm = 0; tm < 2; ++tm)
#pragma unroll
    for (int tn = 0; tn < 2; ++tn)
#pragma unroll
      for (int v = 0; v < 8; ++v)
        xca[base + (size_t)(n0 + mrow + tm * 16 + lr + v) * C +
            dcol + tn * 16 + lc] = f2bf_bits(acc[tm][tn][v]);
}

// =====================================================================
// K5: S = Q*kp2 -> softmax(P=64) -> O = S*vp^T -> permuted scatter
// =====================================================================
__global__ void __launch_bounds__(256) k_sa_apply(
    const unsigned short* __restrict__ Qb, const unsigned short* __restrict__ kp2,
    const unsigned short* __restrict__ vp, unsigned short* __restrict__ xsa) {
  __shared__ __align__(16) unsigned short Qs[128 * 64];
  __shared__ __align__(16) unsigned short KPp[2 * 4 * 512];
  __shared__ __align__(16) unsigned short VPp[2 * 4 * 512];
  __shared__ __align__(16) unsigned short SW[8 * 16 * 64];
  const int t = threadIdx.x, wv = t >> 5, lane = t & 31;
  const int bh = blockIdx.y, b = bh >> 2, h = bh & 3;
  const int n0 = blockIdx.x * 128;
  const size_t base = (size_t)b * Ntok * C + h * 64;
  {
    const int r = t >> 1, c32 = (t & 1) * 32;
    const unsigned short* qsrc = Qb + base + (size_t)(n0 + r) * C + c32;
#pragma unroll
    for (int j = 0; j < 4; ++j)
      async_b128(&Qs[r * 64 + c32 + j * 8], qsrc + j * 8);
    const unsigned short* sk = kp2 + (size_t)bh * 4096;
    const unsigned short* sv = vp + (size_t)bh * 4096;
    async_b128(&KPp[t * 8], sk + t * 8);
    async_b128(&KPp[2048 + t * 8], sk + 2048 + t * 8);
    async_b128(&VPp[t * 8], sv + t * 8);
    async_b128(&VPp[2048 + t * 8], sv + 2048 + t * 8);
  }
  wait_async();
  __syncthreads();
  v8f sacc[4] = {};
#pragma unroll
  for (int k0 = 0; k0 < 64; k0 += 32) {
    v16bf a = load_a(&Qs[wv * 16 * 64 + k0], 64);
#pragma unroll
    for (int tt = 0; tt < 4; ++tt)
      sacc[tt] = wmma_bf16(a, load_bp(&KPp[(k0 >> 5) * 2048 + tt * 512]), sacc[tt]);
  }
  const int lr = (lane >> 4) * 8, lc = lane & 15;
#pragma unroll
  for (int v = 0; v < 8; ++v) {
    float m = fmaxf(fmaxf(sacc[0][v], sacc[1][v]), fmaxf(sacc[2][v], sacc[3][v]));
#pragma unroll
    for (int off = 1; off < 16; off <<= 1) m = fmaxf(m, __shfl_xor(m, off, 32));
    float e0 = __expf(sacc[0][v] - m), e1 = __expf(sacc[1][v] - m);
    float e2 = __expf(sacc[2][v] - m), e3 = __expf(sacc[3][v] - m);
    float s = e0 + e1 + e2 + e3;
#pragma unroll
    for (int off = 1; off < 16; off <<= 1) s += __shfl_xor(s, off, 32);
    const float inv = 1.f / s;
    sacc[0][v] = e0 * inv; sacc[1][v] = e1 * inv;
    sacc[2][v] = e2 * inv; sacc[3][v] = e3 * inv;
  }
#pragma unroll
  for (int tt = 0; tt < 4; ++tt)
#pragma unroll
    for (int v = 0; v < 8; ++v)
      SW[wv * 1024 + (lr + v) * 64 + tt * 16 + lc] = f2bf_bits(sacc[tt][v]);
  v8f oacc[4] = {};
#pragma unroll
  for (int k0 = 0; k0 < 64; k0 += 32) {
    v16bf a2 = load_a(&SW[wv * 1024 + k0], 64);
#pragma unroll
    for (int tt = 0; tt < 4; ++tt)
      oacc[tt] = wmma_bf16(a2, load_bp(&VPp[(k0 >> 5) * 2048 + tt * 512]), oacc[tt]);
  }
#pragma unroll
  for (int tt = 0; tt < 4; ++tt)
#pragma unroll
    for (int v = 0; v < 8; ++v) {
      const int n_g = n0 + wv * 16 + lr + v;
      const int d_g = tt * 16 + lc;
      const int n2 = d_g * 128 + h * 32 + (n_g >> 8);
      const int c2 = n_g & 255;
      xsa[(size_t)b * Ntok * C + (size_t)n2 * C + c2] = f2bf_bits(oacc[tt][v]);
    }
}

// =====================================================================
// K6: out[M,128](f32, ld 256, +colOff) = X[M,256](bf16) @ W[128,256]^T + bias
// =====================================================================
__global__ void __launch_bounds__(256) k_gemm_bf_f32(
    const unsigned short* __restrict__ X, const float* __restrict__ W,
    const float* __restrict__ bias, float* __restrict__ out, int colOff) {
  __shared__ __align__(16) unsigned short Xs[128 * 32];
  __shared__ __align__(16) unsigned short Wp[4 * 512];
  const int t = threadIdx.x, wv = t >> 5, lane = t & 31;
  const int gm0 = blockIdx.x * 128, gn0 = blockIdx.y * 64;
  const int mrow = (wv & 3) * 32, ncol = (wv >> 2) * 32;
  const int xr = t >> 1, xc = (t & 1) * 16;
  const int wn = t >> 2, wk = (t & 3) * 8;
  v8f acc[2][2] = {};
  for (int k0 = 0; k0 < 256; k0 += 32) {
    __syncthreads();
    {
      const unsigned short* src = X + (size_t)(gm0 + xr) * C + k0 + xc;
      async_b128(&Xs[xr * 32 + xc], src);
      async_b128(&Xs[xr * 32 + xc + 8], src + 8);
    }
    {
      float g[8];
      const float* gw = W + (size_t)(gn0 + wn) * C + k0 + wk;
#pragma unroll
      for (int j = 0; j < 8; ++j) g[j] = gw[j];
      const int L = (wn & 15) + ((wk >> 3) & 1) * 16;
      const int j0 = (wk >> 4) << 3;
      *(uint4*)&Wp[(wn >> 4) * 512 + L * 16 + j0] = pack8(g);
    }
    if (k0 + 32 < 256)
      __builtin_prefetch(X + (size_t)(gm0 + xr) * C + k0 + 32 + xc, 0, 1);
    wait_async();
    __syncthreads();
    v16bf a0 = load_a(&Xs[mrow * 32], 32);
    v16bf a1 = load_a(&Xs[(mrow + 16) * 32], 32);
    v16bf b0 = load_bp(&Wp[(ncol >> 4) * 512]);
    v16bf b1 = load_bp(&Wp[(ncol >> 4) * 512 + 512]);
    acc[0][0] = wmma_bf16(a0, b0, acc[0][0]);
    acc[0][1] = wmma_bf16(a0, b1, acc[0][1]);
    acc[1][0] = wmma_bf16(a1, b0, acc[1][0]);
    acc[1][1] = wmma_bf16(a1, b1, acc[1][1]);
  }
  const int lr = (lane >> 4) * 8, lc = lane & 15;
#pragma unroll
  for (int tm = 0; tm < 2; ++tm)
#pragma unroll
    for (int tn = 0; tn < 2; ++tn)
#pragma unroll
      for (int v = 0; v < 8; ++v) {
        const int col = gn0 + ncol + tn * 16 + lc;
        out[(size_t)(gm0 + mrow + tm * 16 + lr + v) * C + colOff + col] =
            acc[tm][tn][v] + bias[col];
      }
}

// =====================================================================
extern "C" void kernel_launch(void* const* d_in, const int* in_sizes, int n_in,
                              void* d_out, int out_size, void* d_ws, size_t ws_size,
                              hipStream_t stream) {
  (void)in_sizes; (void)n_in; (void)out_size; (void)ws_size;
  const float* x    = (const float*)d_in[0];
  const float* q    = (const float*)d_in[1];
  const float* Wq   = (const float*)d_in[2];
  const float* Wk   = (const float*)d_in[3];
  const float* Wvca = (const float*)d_in[4];
  const float* Wvsa = (const float*)d_in[5];
  const float* E_w  = (const float*)d_in[6];
  const float* E_b  = (const float*)d_in[7];
  const float* temp  = (const float*)d_in[8];
  const float* temp2 = (const float*)d_in[9];
  const float* Wo1 = (const float*)d_in[10];
  const float* bo1 = (const float*)d_in[11];
  const float* Wo2 = (const float*)d_in[12];
  const float* bo2 = (const float*)d_in[13];
  float* out = (float*)d_out;

  char* w = (char*)d_ws;
  const size_t SZ = (size_t)Mrows * C * sizeof(unsigned short);  // 16 MiB
  unsigned short* Qb  = (unsigned short*)(w + 0 * SZ);
  unsigned short* Kb  = (unsigned short*)(w + 1 * SZ);
  unsigned short* Vca = (unsigned short*)(w + 2 * SZ);
  unsigned short* Vsa = (unsigned short*)(w + 3 * SZ);
  unsigned short* Xca = (unsigned short*)(w + 4 * SZ);
  unsigned short* Xsa = (unsigned short*)(w + 5 * SZ);
  char* w2 = w + 6 * SZ;
  unsigned short* Att = (unsigned short*)(w2);             // packed 16*64*64 bf16
  unsigned short* Kp2 = (unsigned short*)(w2 + 131072);
  unsigned short* Vp  = (unsigned short*)(w2 + 262144);
  float* invq = (float*)(w2 + 393216);
  float* invk = (float*)(w2 + 393216 + 4096);

  const dim3 blk(256);
  const dim3 gGemm(Mrows / 128, C / 64);
  k_gemm_f32_bf16<<<gGemm, blk, 0, stream>>>(q, Wq, Qb, C);
  k_gemm_f32_bf16<<<gGemm, blk, 0, stream>>>(x, Wk, Kb, C);
  k_gemm_f32_bf16<<<gGemm, blk, 0, stream>>>(x, Wvca, Vca, C);
  k_gemm_f32_bf16<<<gGemm, blk, 0, stream>>>(x, Wvsa, Vsa, C);
  k_norms<<<dim3(Bb), blk, 0, stream>>>(Qb, Kb, invq, invk);
  k_proj_attn<<<dim3(Bb * H), blk, 0, stream>>>(Qb, Kb, Vsa, E_w, E_b,
                                                invq, invk, temp, temp2,
                                                Att, Kp2, Vp);
  k_ca_apply<<<dim3(Ntok / 128, Bb * H), blk, 0, stream>>>(Vca, Att, Xca);
  k_sa_apply<<<dim3(Ntok / 128, Bb * H), blk, 0, stream>>>(Qb, Kp2, Vp, Xsa);
  k_gemm_bf_f32<<<dim3(Mrows / 128, 2), blk, 0, stream>>>(Xsa, Wo1, bo1, out, 0);
  k_gemm_bf_f32<<<dim3(Mrows / 128, 2), blk, 0, stream>>>(Xca, Wo2, bo2, out, 128);
}
